// TransformerBlock_22050362097928
// MI455X (gfx1250) — matrix operand order repaired and checked by reading, not compiled, not run on hardware
//
#include <hip/hip_runtime.h>
#include <math.h>

// ---------------- problem constants ----------------
constexpr int D_  = 2560;
constexpr int H_  = 64;
constexpr int HD_ = 40;
constexpr int T_  = 128;     // encoder tokens
constexpr int S_  = 1152;    // total tokens (128 + 1024)
constexpr int FF_ = 4 * D_;  // 10240
constexpr int TE_ = 512;
constexpr float EPS_ = 1e-5f;
constexpr int G_  = 8;       // heads per batched attention launch

// ---------------- types for WMMA ----------------
typedef __attribute__((ext_vector_type(16))) __bf16  v16bf;
typedef __attribute__((ext_vector_type(8)))  float   v8f;
typedef __attribute__((ext_vector_type(4)))  unsigned int u32x4;

union Frag16 { v16bf v; u32x4 u[2]; };

static __device__ __forceinline__ unsigned short f2bf(float f) {
  unsigned int u = __float_as_uint(f);
  u += 0x7fffu + ((u >> 16) & 1u);          // round-to-nearest-even
  return (unsigned short)(u >> 16);
}

// =====================================================================
// bf16 WMMA GEMM, batched over blockIdx.z with element strides.
//   C(MxN) = alpha * A(MxK) @ Bt(NxK)^T + bias     (f32 or bf16 store)
// A row-major (lda), Bt row-major N-by-K (ldb).
// Requirements: K % 32 == 0; lda/ldb multiples of 8; bases 16B aligned;
//               M % 16 == 0.
// Block = 256 threads = 8 waves arranged 4(M) x 2(N-group).
// Each wave computes a 16x64 strip: 1 A fragment reused by 4 WMMAs.
// =====================================================================
__global__ void gemm_bf16_kernel(const unsigned short* __restrict__ A, int lda, size_t strideA,
                                 const unsigned short* __restrict__ Bt, int ldb, size_t strideB,
                                 const float* __restrict__ bias,
                                 void* __restrict__ Cv, int ldc, size_t strideC,
                                 int M, int N, int K, int n_store,
                                 float alpha, int store_bf16)
{
  const int lane = threadIdx.x & 31;
  const int wave = threadIdx.x >> 5;
  const int tileM = blockIdx.y * 4 + (wave & 3);
  const int nbase = (blockIdx.x * 2 + (wave >> 2)) * 64;
  if (tileM * 16 >= M || nbase >= N) return;   // wave-uniform

  const int r  = lane & 15;
  const int hs = lane >> 4;   // half-select
  const int aoff0 = hs * 16;       // elems 0..7  -> K = kb + aoff0 + e
  const int aoff1 = hs * 16 + 8;   // elems 8..15 -> K = kb + aoff1 + (e-8)
  const int boff  = hs * 16;       // elems 0..15 -> K = kb + boff + e

  const unsigned short* Ab = A  + blockIdx.z * strideA;
  const unsigned short* Bb = Bt + blockIdx.z * strideB;
  const unsigned short* Aptr = Ab + (size_t)(tileM * 16 + r) * lda;

  const unsigned short* Bp[4];
  #pragma unroll
  for (int j = 0; j < 4; ++j) {
    int cb = nbase + j * 16 + r;
    if (cb > N - 1) cb = N - 1;          // clamp (values discarded at store)
    Bp[j] = Bb + (size_t)cb * ldb;
  }

  v8f acc[4] = {v8f{}, v8f{}, v8f{}, v8f{}};
  for (int kb = 0; kb < K; kb += 32) {
    Frag16 a;
    a.u[0] = *(const u32x4*)(Aptr + kb + aoff0);
    a.u[1] = *(const u32x4*)(Aptr + kb + aoff1);
    Frag16 b[4];
    #pragma unroll
    for (int j = 0; j < 4; ++j) {
      b[j].u[0] = *(const u32x4*)(Bp[j] + kb + boff);
      b[j].u[1] = *(const u32x4*)(Bp[j] + kb + boff + 8);
    }
    #pragma unroll
    for (int j = 0; j < 4; ++j)
      acc[j] = __builtin_amdgcn_wmma_f32_16x16x32_bf16(
          false, a.v, false, b[j].v, (short)0, acc[j], false, false);
  }

  #pragma unroll
  for (int j = 0; j < 4; ++j) {
    const int col = nbase + j * 16 + r;
    if (col >= n_store) continue;
    const float bvv = bias ? bias[col] : 0.f;
    if (!store_bf16) {
      float* C = (float*)Cv + blockIdx.z * strideC;
      #pragma unroll
      for (int i = 0; i < 8; ++i) {
        int row = tileM * 16 + i + hs * 8;
        C[(size_t)row * ldc + col] = alpha * acc[j][i] + bvv;
      }
    } else {
      unsigned short* C = (unsigned short*)Cv + blockIdx.z * strideC;
      #pragma unroll
      for (int i = 0; i < 8; ++i) {
        int row = tileM * 16 + i + hs * 8;
        C[(size_t)row * ldc + col] = f2bf(alpha * acc[j][i] + bvv);
      }
    }
  }
}

// ------------- f32 -> bf16 transpose-convert: out(C,R) = in(R,C)^T -------------
__global__ void tconv_kernel(const float* __restrict__ in, unsigned short* __restrict__ out,
                             int R, int C)
{
  size_t idx = (size_t)blockIdx.x * 256 + threadIdx.x;
  if (idx >= (size_t)R * C) return;
  int c = (int)(idx % C);
  int r = (int)(idx / C);
  out[(size_t)c * R + r] = f2bf(in[idx]);
}

// ------------- AdaLN embedding GEMV: emb = time_embed @ ada_w + ada_b -------------
__global__ void ada_kernel(const float* __restrict__ t, const float* __restrict__ W,
                           const float* __restrict__ b, float* __restrict__ emb)
{
  int j = blockIdx.x * 256 + threadIdx.x;
  if (j >= 12 * D_) return;
  float acc = b[j];
  for (int i = 0; i < TE_; ++i)
    acc += t[i] * W[(size_t)i * (12 * D_) + j];
  emb[j] = acc;
}

// ------------- LayerNorm + AdaLN modulation, writes concat [e;h] as bf16 -------------
__global__ void ln_mod_kernel(const float* __restrict__ src_e, const float* __restrict__ src_h,
                              const float* __restrict__ emb,
                              int scale_e_off, int shift_e_off,
                              int scale_h_off, int shift_h_off,
                              unsigned short* __restrict__ out)
{
  __shared__ float s_sum[256];
  __shared__ float s_sq[256];
  int s = blockIdx.x;
  const float *src, *scale, *shift;
  if (s < T_) { src = src_e + (size_t)s * D_;        scale = emb + scale_e_off; shift = emb + shift_e_off; }
  else        { src = src_h + (size_t)(s - T_) * D_; scale = emb + scale_h_off; shift = emb + shift_h_off; }

  float sum = 0.f, sq = 0.f;
  for (int d = threadIdx.x; d < D_; d += 256) { float v = src[d]; sum += v; sq += v * v; }
  s_sum[threadIdx.x] = sum; s_sq[threadIdx.x] = sq; __syncthreads();
  for (int o = 128; o > 0; o >>= 1) {
    if (threadIdx.x < o) { s_sum[threadIdx.x] += s_sum[threadIdx.x + o]; s_sq[threadIdx.x] += s_sq[threadIdx.x + o]; }
    __syncthreads();
  }
  float mu  = s_sum[0] * (1.f / D_);
  float var = s_sq[0] * (1.f / D_) - mu * mu;
  float rin = rsqrtf(var + EPS_);
  for (int d = threadIdx.x; d < D_; d += 256) {
    float v = (src[d] - mu) * rin;
    out[(size_t)s * D_ + d] = f2bf(v * (1.f + scale[d]) + shift[d]);
  }
}

// ------------- per-head LN (HD=40) + RoPE on rows >= T, pad K to 64, bf16 -------------
__global__ void qk_ln_rope_kernel(const float* __restrict__ qf, const float* __restrict__ kf,
                                  const float* __restrict__ cosb, const float* __restrict__ sinb,
                                  unsigned short* __restrict__ qp, unsigned short* __restrict__ kp)
{
  int s = blockIdx.x, h = blockIdx.y;
  const float* in = blockIdx.z ? kf : qf;
  unsigned short* out = blockIdx.z ? kp : qp;
  int t = threadIdx.x;                      // 0..63
  __shared__ float ssum[64], ssq[64], sv[40];
  float v = 0.f;
  if (t < HD_) v = in[(size_t)s * D_ + h * HD_ + t];
  ssum[t] = v; ssq[t] = v * v; __syncthreads();
  for (int o = 32; o > 0; o >>= 1) {
    if (t < o) { ssum[t] += ssum[t + o]; ssq[t] += ssq[t + o]; }
    __syncthreads();
  }
  float mu  = ssum[0] * (1.f / HD_);
  float var = ssq[0] * (1.f / HD_) - mu * mu;
  float rin = rsqrtf(var + EPS_);
  float n = (v - mu) * rin;
  if (t < HD_) sv[t] = n;
  __syncthreads();
  float o_ = n;
  if (t < HD_ && s >= T_) {
    int p = s - T_;
    float c  = cosb[p * HD_ + t];
    float sn = sinb[p * HD_ + t];
    float rot = (t < HD_ / 2) ? -sv[t + HD_ / 2] : sv[t - HD_ / 2];
    o_ = n * c + rot * sn;
  }
  out[((size_t)h * S_ + s) * 64 + t] = (t < HD_) ? f2bf(o_) : (unsigned short)0;
}

// ------------- V transposed per head: vt(H, 48, S) bf16, rows 40..47 zero -------------
__global__ void vt_kernel(const float* __restrict__ vf, unsigned short* __restrict__ vt)
{
  size_t idx = (size_t)blockIdx.x * 256 + threadIdx.x;
  size_t total = (size_t)H_ * 48 * S_;
  if (idx >= total) return;
  int s = (int)(idx % S_);
  size_t r = idx / S_;
  int n = (int)(r % 48);
  int h = (int)(r / 48);
  float val = (n < HD_) ? vf[(size_t)s * D_ + h * HD_ + n] : 0.f;
  vt[idx] = f2bf(val);
}

// ------------- row softmax (f32 in) -> bf16 probabilities, batched over grid.y -------------
__global__ void softmax_kernel(const float* __restrict__ sc, unsigned short* __restrict__ pb)
{
  __shared__ float sh[256];
  size_t row = (size_t)blockIdx.y * S_ + blockIdx.x;
  const float* x = sc + row * S_;
  float m = -1e30f;
  for (int j = threadIdx.x; j < S_; j += 256) m = fmaxf(m, x[j]);
  sh[threadIdx.x] = m; __syncthreads();
  for (int o = 128; o > 0; o >>= 1) {
    if (threadIdx.x < o) sh[threadIdx.x] = fmaxf(sh[threadIdx.x], sh[threadIdx.x + o]);
    __syncthreads();
  }
  m = sh[0]; __syncthreads();
  float sum = 0.f;
  for (int j = threadIdx.x; j < S_; j += 256) sum += __expf(x[j] - m);
  sh[threadIdx.x] = sum; __syncthreads();
  for (int o = 128; o > 0; o >>= 1) {
    if (threadIdx.x < o) sh[threadIdx.x] += sh[threadIdx.x + o];
    __syncthreads();
  }
  float inv = 1.f / sh[0];
  for (int j = threadIdx.x; j < S_; j += 256)
    pb[row * S_ + j] = f2bf(__expf(x[j] - m) * inv);
}

// ------------- first residual with gates: res = input + attn_out * gate -------------
__global__ void resgate_kernel(const float* __restrict__ hs, const float* __restrict__ es,
                               const float* __restrict__ ao, const float* __restrict__ emb,
                               float* __restrict__ res)
{
  size_t idx = (size_t)blockIdx.x * 256 + threadIdx.x;
  if (idx >= (size_t)S_ * D_) return;
  int d = (int)(idx % D_);
  int s = (int)(idx / D_);
  float g, base;
  if (s < T_) { g = emb[5 * D_ + d]; base = es[(size_t)s * D_ + d]; }                 // c_gate_msa
  else        { g = emb[4 * D_ + d]; base = hs[(size_t)(s - T_) * D_ + d]; }          // gate_msa
  res[idx] = base + ao[idx] * g;
}

// ------------- tanh GELU, f32 -> bf16 -------------
__global__ void gelu_kernel(const float* __restrict__ in, unsigned short* __restrict__ out, size_t n)
{
  size_t idx = (size_t)blockIdx.x * 256 + threadIdx.x;
  if (idx >= n) return;
  float x = in[idx];
  float t = tanhf(0.7978845608028654f * (x + 0.044715f * x * x * x));
  out[idx] = f2bf(0.5f * x * (1.f + t));
}

// ------------- final residual + MLP gates, scatter to output layout -------------
__global__ void final_kernel(const float* __restrict__ res, const float* __restrict__ ff2,
                             const float* __restrict__ emb, float* __restrict__ out)
{
  size_t idx = (size_t)blockIdx.x * 256 + threadIdx.x;
  if (idx >= (size_t)S_ * D_) return;
  int d = (int)(idx % D_);
  int s = (int)(idx / D_);
  float g = (s < T_) ? emb[11 * D_ + d] : emb[10 * D_ + d];  // c_gate_mlp / gate_mlp
  float v = res[idx] + ff2[idx] * g;
  if (s < T_) out[(size_t)(S_ - T_) * D_ + (size_t)s * D_ + d] = v;   // encoder after hidden
  else        out[(size_t)(s - T_) * D_ + d] = v;                     // hidden first
}

// =====================================================================
extern "C" void kernel_launch(void* const* d_in, const int* in_sizes, int n_in,
                              void* d_out, int out_size, void* d_ws, size_t ws_size,
                              hipStream_t stream)
{
  (void)in_sizes; (void)n_in; (void)out_size; (void)ws_size;
  const float* hs   = (const float*)d_in[0];
  const float* es   = (const float*)d_in[1];
  const float* te   = (const float*)d_in[2];
  const float* rc   = (const float*)d_in[3];
  const float* rs   = (const float*)d_in[4];
  const float* adaw = (const float*)d_in[5];
  const float* adab = (const float*)d_in[6];
  const float* wq = (const float*)d_in[7];  const float* bq = (const float*)d_in[8];
  const float* wk = (const float*)d_in[9];  const float* bk = (const float*)d_in[10];
  const float* wv = (const float*)d_in[11]; const float* bvp = (const float*)d_in[12];
  const float* wo = (const float*)d_in[13]; const float* bo = (const float*)d_in[14];
  const float* w1 = (const float*)d_in[15]; const float* b1 = (const float*)d_in[16];
  const float* w2 = (const float*)d_in[17]; const float* b2 = (const float*)d_in[18];

  char* p = (char*)d_ws;
  auto carve = [&](size_t bytes) -> char* {
    char* r = p; p += (bytes + 255) & ~(size_t)255; return r;
  };
  unsigned short* wq_t = (unsigned short*)carve((size_t)D_ * D_ * 2);
  unsigned short* wk_t = (unsigned short*)carve((size_t)D_ * D_ * 2);
  unsigned short* wv_t = (unsigned short*)carve((size_t)D_ * D_ * 2);
  unsigned short* wo_t = (unsigned short*)carve((size_t)D_ * D_ * 2);
  unsigned short* w1_t = (unsigned short*)carve((size_t)D_ * FF_ * 2);
  unsigned short* w2_t = (unsigned short*)carve((size_t)FF_ * D_ * 2);
  float*          emb  = (float*)carve((size_t)12 * D_ * 4);
  unsigned short* xb   = (unsigned short*)carve((size_t)S_ * D_ * 2);
  float* qf = (float*)carve((size_t)S_ * D_ * 4);
  float* kf = (float*)carve((size_t)S_ * D_ * 4);
  float* vf = (float*)carve((size_t)S_ * D_ * 4);
  unsigned short* qp = (unsigned short*)carve((size_t)H_ * S_ * 64 * 2);
  unsigned short* kp = (unsigned short*)carve((size_t)H_ * S_ * 64 * 2);
  unsigned short* vt = (unsigned short*)carve((size_t)H_ * 48 * S_ * 2);
  float*          sc = (float*)carve((size_t)G_ * S_ * S_ * 4);
  unsigned short* pb = (unsigned short*)carve((size_t)G_ * S_ * S_ * 2);
  unsigned short* ob = (unsigned short*)carve((size_t)S_ * D_ * 2);
  float* ao  = (float*)carve((size_t)S_ * D_ * 4);
  float* res = (float*)carve((size_t)S_ * D_ * 4);
  float* ff1 = (float*)carve((size_t)S_ * FF_ * 4);
  unsigned short* g1 = (unsigned short*)carve((size_t)S_ * FF_ * 2);
  float* ff2 = (float*)carve((size_t)S_ * D_ * 4);

  auto tconv = [&](const float* in, unsigned short* out, int R, int C) {
    size_t total = (size_t)R * C;
    tconv_kernel<<<dim3((unsigned)((total + 255) / 256)), dim3(256), 0, stream>>>(in, out, R, C);
  };
  auto gemm = [&](const unsigned short* A, int lda, size_t sA,
                  const unsigned short* Bt, int ldb, size_t sB,
                  const float* bias, void* C, int ldc, size_t sC,
                  int M, int N, int K, int n_store, float alpha, int store_bf16, int batch) {
    dim3 grid((N + 127) / 128, (M + 63) / 64, batch);
    gemm_bf16_kernel<<<grid, dim3(256), 0, stream>>>(A, lda, sA, Bt, ldb, sB, bias, C, ldc, sC,
                                                     M, N, K, n_store, alpha, store_bf16);
  };

  // 1) weights -> bf16, transposed to (N,K)
  tconv(wq, wq_t, D_, D_);
  tconv(wk, wk_t, D_, D_);
  tconv(wv, wv_t, D_, D_);
  tconv(wo, wo_t, D_, D_);
  tconv(w1, w1_t, D_, FF_);   // w1 (D,FF) -> (FF,D)
  tconv(w2, w2_t, FF_, D_);   // w2 (FF,D) -> (D,FF)

  // 2) AdaLN embedding
  ada_kernel<<<dim3((12 * D_ + 255) / 256), dim3(256), 0, stream>>>(te, adaw, adab, emb);

  // 3) LN + MSA modulation -> x_bf16 (concat [enc; hid])
  //    chunks: 0 shift_msa, 1 c_shift_msa, 2 scale_msa, 3 c_scale_msa
  ln_mod_kernel<<<dim3(S_), dim3(256), 0, stream>>>(es, hs, emb,
                                                    3 * D_, 1 * D_, 2 * D_, 0 * D_, xb);

  // 4) QKV projections (WMMA)
  gemm(xb, D_, 0, wq_t, D_, 0, bq,  qf, D_, 0, S_, D_, D_, D_, 1.f, 0, 1);
  gemm(xb, D_, 0, wk_t, D_, 0, bk,  kf, D_, 0, S_, D_, D_, D_, 1.f, 0, 1);
  gemm(xb, D_, 0, wv_t, D_, 0, bvp, vf, D_, 0, S_, D_, D_, D_, 1.f, 0, 1);

  // 5) per-head LN + RoPE, pad HD 40 -> 64 (bf16)
  qk_ln_rope_kernel<<<dim3(S_, H_, 2), dim3(64), 0, stream>>>(qf, kf, rc, rs, qp, kp);
  {
    size_t total = (size_t)H_ * 48 * S_;
    vt_kernel<<<dim3((unsigned)((total + 255) / 256)), dim3(256), 0, stream>>>(vf, vt);
  }

  // 6) attention, 8 heads per batched launch:
  //    scores (WMMA) -> softmax -> P@V (WMMA, bf16 out)
  const float inv_sqrt_hd = 0.15811388300841897f;   // 1/sqrt(40)
  for (int g0 = 0; g0 < H_; g0 += G_) {
    gemm(qp + (size_t)g0 * S_ * 64, 64, (size_t)S_ * 64,
         kp + (size_t)g0 * S_ * 64, 64, (size_t)S_ * 64,
         nullptr, sc, S_, (size_t)S_ * S_,
         S_, S_, 64, S_, inv_sqrt_hd, 0, G_);
    softmax_kernel<<<dim3(S_, G_), dim3(256), 0, stream>>>(sc, pb);
    gemm(pb, S_, (size_t)S_ * S_,
         vt + (size_t)g0 * 48 * S_, S_, (size_t)48 * S_,
         nullptr, ob + (size_t)g0 * HD_, D_, (size_t)HD_,
         S_, 48, S_, HD_, 1.f, 1, G_);
  }

  // 7) output projection (WMMA)
  gemm(ob, D_, 0, wo_t, D_, 0, bo, ao, D_, 0, S_, D_, D_, D_, 1.f, 0, 1);

  // 8) gated residual #1
  {
    size_t total = (size_t)S_ * D_;
    resgate_kernel<<<dim3((unsigned)((total + 255) / 256)), dim3(256), 0, stream>>>(hs, es, ao, emb, res);
  }

  // 9) LN + MLP modulation (chunks: 6 shift_mlp, 7 c_shift_mlp, 8 scale_mlp, 9 c_scale_mlp)
  ln_mod_kernel<<<dim3(S_), dim3(256), 0, stream>>>(res, res + (size_t)T_ * D_, emb,
                                                    9 * D_, 7 * D_, 8 * D_, 6 * D_, xb);

  // 10) MLP (WMMA): ff1 -> GELU(bf16) -> ff2
  gemm(xb, D_, 0, w1_t, D_, 0, b1, ff1, FF_, 0, S_, FF_, D_, FF_, 1.f, 0, 1);
  {
    size_t total = (size_t)S_ * FF_;
    gelu_kernel<<<dim3((unsigned)((total + 255) / 256)), dim3(256), 0, stream>>>(ff1, g1, total);
  }
  gemm(g1, FF_, 0, w2_t, FF_, 0, b2, ff2, D_, 0, S_, D_, FF_, D_, 1.f, 0, 1);

  // 11) final gated residual, write [hidden(1024,D); encoder(128,D)] to d_out
  {
    size_t total = (size_t)S_ * D_;
    final_kernel<<<dim3((unsigned)((total + 255) / 256)), dim3(256), 0, stream>>>(res, ff2, emb, (float*)d_out);
  }
}
